// EncoderRNN_74483322847540
// MI455X (gfx1250) — matrix operand order repaired and checked by reading, not compiled, not run on hardware
//
#include <hip/hip_runtime.h>
#include <hip/hip_bf16.h>
#include <math.h>

typedef _Float16 f16;
typedef __attribute__((ext_vector_type(16))) _Float16 v16h;
typedef __attribute__((ext_vector_type(8)))  _Float16 v8h;
typedef __attribute__((ext_vector_type(8)))  float    v8f;
typedef __attribute__((ext_vector_type(4)))  unsigned int u32x4;
typedef __attribute__((ext_vector_type(8)))  int i32x8;
typedef __attribute__((ext_vector_type(4)))  int i32x4;

#define B_  8
#define S_  1024
#define FIN 128
#define E_  256
#define H_  256
#define L_  2
#define HEADS_ 4
#define KD_ 64
#define BS_ (B_*S_)   // 8192

// ---------------------------------------------------------------------------
// Generic f16 WMMA GEMM: C[M,N] = A[M,K] x B[K,N], f32 accumulate.
// Wave computes a 16x16 tile; block = 4 waves stacked in M (64x16 tile).
// Requires M%64==0, N%16==0, K%32==0 (all shapes here satisfy this).
// Epilogue: +bias, +posemb(row%S), relu, *rowmask, +resid; f32/f16/f16-transposed.
// ---------------------------------------------------------------------------
__global__ __launch_bounds__(128) void k_gemm(
    const f16* __restrict__ A, const f16* __restrict__ Bm,
    const float* __restrict__ bias, const float* __restrict__ pos,
    const unsigned char* __restrict__ rowmask, const float* __restrict__ resid,
    float* __restrict__ C, f16* __restrict__ C16, f16* __restrict__ C16t,
    int M, int N, int K, int act)
{
  const int lane = threadIdx.x & 31;
  const int wv   = threadIdx.x >> 5;
  const int l16  = lane & 15;
  const int hsel = lane >> 4;
  const int tileM = blockIdx.x * 64 + wv * 16;
  const int n0    = blockIdx.y * 16;
  const f16* arow = A + (size_t)(tileM + l16) * K;

  v8f acc = {};
  for (int k0 = 0; k0 < K; k0 += 32) {
    // A 16x32: lane = m, slots 0..7 -> K = k0+8*hsel+i ; slots 8..15 -> +16
    v8h alo = *(const v8h*)(arow + k0 + 8 * hsel);
    v8h ahi = *(const v8h*)(arow + k0 + 16 + 8 * hsel);
    // B 32x16: lane = n (n0+l16), slots = K same order as A
    const f16* bp = Bm + (size_t)(k0 + 8 * hsel) * N + n0 + l16;
    v8h blo, bhi;
#pragma unroll
    for (int i = 0; i < 8; ++i) { blo[i] = bp[(size_t)i * N]; bhi[i] = bp[(size_t)(16 + i) * N]; }
    v16h a, b;
#pragma unroll
    for (int i = 0; i < 8; ++i) { a[i] = alo[i]; a[8+i] = ahi[i]; b[i] = blo[i]; b[8+i] = bhi[i]; }
    acc = __builtin_amdgcn_wmma_f32_16x16x32_f16(false, a, false, b, (short)0, acc, false, false);
  }

#pragma unroll
  for (int r = 0; r < 8; ++r) {
    // C/D layout: vgpr r -> m = r + 8*hsel ; lane -> n = l16
    int gm = tileM + hsel * 8 + r;
    int gn = n0 + l16;
    float v = acc[r];
    if (bias)    v += bias[gn];
    if (pos)     v += pos[(size_t)(gm & (S_ - 1)) * N + gn];
    if (act)     v = v > 0.f ? v : 0.f;
    if (rowmask) v *= rowmask[gm] ? 1.f : 0.f;
    if (resid)   v += resid[(size_t)gm * N + gn];
    size_t o = (size_t)gm * N + gn;
    if (C)    C[o]   = v;
    if (C16)  C16[o] = (f16)v;
    if (C16t) C16t[((size_t)(gm >> 10) * (HEADS_*KD_) + gn) * S_ + (gm & (S_ - 1))] = (f16)v;
  }
}

// ---------------------------------------------------------------------------
// LSTM recurrent scan for one layer. One block (16 waves, 512 threads) holds
// h/c state in LDS; per step: TDM DMAs the zx[:,t,:] tile into LDS while all
// waves compute h @ Wh with WMMA; then zbuf += wmma results, gate update.
// Gate order i,f,g,o. Output hs[b,t,j] is the pre-mask h_t.
// ---------------------------------------------------------------------------
__global__ __launch_bounds__(512) void k_lstm_scan(
    const float* __restrict__ zx,   // [B,S,4H]
    const f16*   __restrict__ Wh,   // [H,4H] f16
    float*       __restrict__ hs)   // [B,S,H]
{
  __shared__ f16   h16[16 * H_];      // rows 8..15 stay zero (WMMA M pad)
  __shared__ float cst[B_ * H_];
  __shared__ float zbuf[B_ * 4 * H_];

  const int tid  = threadIdx.x;
  const int lane = tid & 31;
  const int wv   = tid >> 5;
  const int l16  = lane & 15;
  const int hsel = lane >> 4;

  for (int i = tid; i < 16 * H_; i += 512) h16[i] = (f16)0.f;
  for (int i = tid; i < B_ * H_; i += 512) cst[i] = 0.f;
  __syncthreads();

  for (int t = 0; t < S_; ++t) {
#if __has_builtin(__builtin_amdgcn_tensor_load_to_lds)
    // TDM: stage zx[:,t,:] (8 rows x 4KB, row stride S*4H floats) into zbuf.
    // Issued by wave 0 only (TDM ignores EXEC); overlaps the WMMAs below.
    if (wv == 0) {
      unsigned lds_base = (unsigned)(uintptr_t)&zbuf[0];
      unsigned long long ga = (unsigned long long)(uintptr_t)&zx[(size_t)t * (4 * H_)];
      u32x4 g0;
      g0[0] = 1u;                                         // count=1, user descriptor
      g0[1] = lds_base;                                   // lds_addr
      g0[2] = (unsigned)(ga & 0xffffffffu);               // global_addr[31:0]
      g0[3] = (unsigned)((ga >> 32) & 0x1ffffffu) | (2u << 30);  // addr[56:32] | type=2
      i32x8 g1;
      g1[0] = (int)(2u << 16);                            // data_size = 4 bytes
      g1[1] = (int)((4 * H_) << 16);                      // tensor_dim0[15:0] = 1024
      g1[2] = (int)(B_ << 16);                            // dim0 hi=0 | tensor_dim1[15:0] = 8
      g1[3] = (int)((4 * H_) << 16);                      // dim1 hi=0 | tile_dim0 = 1024
      g1[4] = (int)B_;                                    // tile_dim1 = 8, tile_dim2 = 0
      g1[5] = (int)(S_ * 4 * H_);                         // tensor_dim0_stride[31:0] = 2^20
      g1[6] = 0;                                          // stride hi | dim1_stride lo
      g1[7] = 0;
      i32x4 gz4 = {0, 0, 0, 0};
      i32x8 gz8 = {0, 0, 0, 0, 0, 0, 0, 0};
      __builtin_amdgcn_tensor_load_to_lds(g0, g1, gz4, gz4, gz8, 0);
    }
#else
    for (int c = tid; c < B_ * H_; c += 512) {            // fallback: plain copy
      int bb = c >> 8, n4 = (c & 255) * 4;
      *(float4*)&zbuf[bb * 4 * H_ + n4] =
          *(const float4*)&zx[((size_t)bb * S_ + t) * (4 * H_) + n4];
    }
#endif
    v8f accs[4];
#pragma unroll
    for (int nt = 0; nt < 4; ++nt) {
      const int n0 = (wv * 4 + nt) * 16;
      v8f acc = {};
      for (int k0 = 0; k0 < H_; k0 += 32) {
        v8h alo = *(const v8h*)&h16[l16 * H_ + k0 + 8 * hsel];
        v8h ahi = *(const v8h*)&h16[l16 * H_ + k0 + 16 + 8 * hsel];
        const f16* bp = Wh + (size_t)(k0 + 8 * hsel) * (4 * H_) + n0 + l16;
        v8h blo, bhi;
#pragma unroll
        for (int i = 0; i < 8; ++i) { blo[i] = bp[(size_t)i * (4 * H_)]; bhi[i] = bp[(size_t)(16 + i) * (4 * H_)]; }
        v16h a, b;
#pragma unroll
        for (int i = 0; i < 8; ++i) { a[i]=alo[i]; a[8+i]=ahi[i]; b[i]=blo[i]; b[8+i]=bhi[i]; }
        acc = __builtin_amdgcn_wmma_f32_16x16x32_f16(false, a, false, b, (short)0, acc, false, false);
      }
      accs[nt] = acc;
    }
#if __has_builtin(__builtin_amdgcn_tensor_load_to_lds)
    __builtin_amdgcn_s_wait_tensorcnt(0);   // wave 0 waits its DMA; others no-op
#endif
    __syncthreads();                        // zbuf (zx tile) visible to all waves
#pragma unroll
    for (int nt = 0; nt < 4; ++nt) {
      const int n0 = (wv * 4 + nt) * 16;
#pragma unroll
      for (int r = 0; r < 8; ++r) {
        int bb = hsel * 8 + r;              // tile row = batch index
        if (bb < B_) zbuf[bb * 4 * H_ + n0 + l16] += accs[nt][r];
      }
    }
    __syncthreads();
    for (int u = tid; u < B_ * H_; u += 512) {
      int b = u >> 8, j = u & (H_ - 1);
      const float* z = &zbuf[b * 4 * H_];
      float zi = z[j], zf = z[H_ + j], zg = z[2 * H_ + j], zo = z[3 * H_ + j];
      float si = 1.f / (1.f + expf(-zi));
      float sf = 1.f / (1.f + expf(-zf));
      float so = 1.f / (1.f + expf(-zo));
      float c  = sf * cst[u] + si * tanhf(zg);
      float h  = so * tanhf(c);
      cst[u] = c;
      h16[b * H_ + j] = (f16)h;
      hs[((size_t)b * S_ + t) * H_ + j] = h;
    }
    __syncthreads();
  }
}

// ---------------------------------------------------------------------------
// WMMA flash attention. Block = 4 waves, each wave owns 16 query rows of one
// (b, head). Keys processed 32 at a time: scores = Q x K^T via WMMA, online
// softmax (row stats via 16-lane shfl_xor reductions), P goes through a
// per-wave LDS tile (C-layout -> A-layout), then O += P x V via WMMA.
// ---------------------------------------------------------------------------
__global__ __launch_bounds__(128) void k_attn_wmma(
    const f16* __restrict__ Qp,   // [BS, HEADS*KD] f16
    const f16* __restrict__ Kt,   // [B, HEADS*KD, S] f16 (transposed)
    const f16* __restrict__ Vp,   // [BS, HEADS*KD] f16
    const int* __restrict__ lengths,
    f16* __restrict__ out16)      // [BS, HEADS*KD]
{
  __shared__ f16 Pt[4][16 * 32];
  const int lane = threadIdx.x & 31;
  const int wv   = threadIdx.x >> 5;
  const int l16  = lane & 15;
  const int hsel = lane >> 4;
  const int bh   = blockIdx.x >> 4;           // grid.x = B*HEADS*(S/64)
  const int qblk = blockIdx.x & 15;
  const int b = bh >> 2, hd = bh & 3;
  const int q0 = qblk * 64 + wv * 16;
  const int len = lengths[b];

  // Resident Q A-operand fragments (feature chunks 0..31 and 32..63)
  const f16* qrow = Qp + ((size_t)(b * S_) + q0 + l16) * (HEADS_*KD_) + hd * KD_;
  v16h qa[2];
#pragma unroll
  for (int c = 0; c < 2; ++c) {
    v8h lo = *(const v8h*)(qrow + c * 32 + 8 * hsel);
    v8h hi = *(const v8h*)(qrow + c * 32 + 16 + 8 * hsel);
#pragma unroll
    for (int i = 0; i < 8; ++i) { qa[c][i] = lo[i]; qa[c][8+i] = hi[i]; }
  }

  float mrow[8], lrow[8];
  v8f O[4];
#pragma unroll
  for (int r = 0; r < 8; ++r) { mrow[r] = -3.0e38f; lrow[r] = 0.f; }
#pragma unroll
  for (int j = 0; j < 4; ++j) O[j] = (v8f){};

  const int nkb = (len + 31) >> 5;
  for (int kb32 = 0; kb32 < nkb; ++kb32) {
    const int kb = kb32 * 32;
    v8f s0 = {}, s1 = {};
#pragma unroll
    for (int c = 0; c < 2; ++c) {
      // B operand: lane = key (kb+l16 / kb+16+l16), slots = features in A's k order
      const f16* ktp = Kt + ((size_t)b * (HEADS_*KD_) + hd * KD_ + c * 32 + 8 * hsel) * S_ + kb + l16;
      v8h b0lo, b0hi, b1lo, b1hi;
#pragma unroll
      for (int i = 0; i < 8; ++i) { b0lo[i] = ktp[(size_t)i * S_]; b0hi[i] = ktp[(size_t)(16 + i) * S_]; b1lo[i] = ktp[(size_t)i * S_ + 16]; b1hi[i] = ktp[(size_t)(16 + i) * S_ + 16]; }
      v16h b0, b1;
#pragma unroll
      for (int i = 0; i < 8; ++i) { b0[i]=b0lo[i]; b0[8+i]=b0hi[i]; b1[i]=b1lo[i]; b1[8+i]=b1hi[i]; }
      s0 = __builtin_amdgcn_wmma_f32_16x16x32_f16(false, qa[c], false, b0, (short)0, s0, false, false);
      s1 = __builtin_amdgcn_wmma_f32_16x16x32_f16(false, qa[c], false, b1, (short)0, s1, false, false);
    }
    // scale + key mask (lane's key index: kb+l16 and kb+16+l16)
    const float f0m = (kb + l16)      < len ? 0.f : -3.0e38f;
    const float f1m = (kb + 16 + l16) < len ? 0.f : -3.0e38f;
#pragma unroll
    for (int r = 0; r < 8; ++r) {
      float a0 = s0[r] * 0.125f + f0m;       // 1/sqrt(64)
      float a1 = s1[r] * 0.125f + f1m;
      float tm = fmaxf(a0, a1);
      tm = fmaxf(tm, __shfl_xor(tm, 1, 32));
      tm = fmaxf(tm, __shfl_xor(tm, 2, 32));
      tm = fmaxf(tm, __shfl_xor(tm, 4, 32));
      tm = fmaxf(tm, __shfl_xor(tm, 8, 32));
      float nm   = fmaxf(mrow[r], tm);
      float corr = __expf(mrow[r] - nm);
      float p0 = __expf(a0 - nm), p1 = __expf(a1 - nm);
      float ts = p0 + p1;
      ts += __shfl_xor(ts, 1, 32);
      ts += __shfl_xor(ts, 2, 32);
      ts += __shfl_xor(ts, 4, 32);
      ts += __shfl_xor(ts, 8, 32);
      lrow[r] = lrow[r] * corr + ts;
      mrow[r] = nm;
#pragma unroll
      for (int j = 0; j < 4; ++j) O[j][r] *= corr;
      int m = r + 8 * hsel;                  // C-layout row -> LDS (A-layout source)
      Pt[wv][m * 32 + l16]      = (f16)p0;
      Pt[wv][m * 32 + 16 + l16] = (f16)p1;
    }
    __asm volatile("s_wait_dscnt 0x0" ::: "memory");
    // P as A-operand (16 q rows x 32 keys) from LDS
    v8h plo = *(const v8h*)&Pt[wv][l16 * 32 + 8 * hsel];
    v8h phi = *(const v8h*)&Pt[wv][l16 * 32 + 16 + 8 * hsel];
    v16h pa;
#pragma unroll
    for (int i = 0; i < 8; ++i) { pa[i] = plo[i]; pa[8+i] = phi[i]; }
    // V B-operand: lane = feature (j*16+l16), slots = keys in A's k order
    const f16* vrow = Vp + ((size_t)(b * S_) + kb + 8 * hsel) * (HEADS_*KD_) + hd * KD_ + l16;
#pragma unroll
    for (int j = 0; j < 4; ++j) {
      v8h vlo, vhi;
#pragma unroll
      for (int i = 0; i < 8; ++i) { vlo[i] = vrow[(size_t)i * (HEADS_*KD_) + j * 16]; vhi[i] = vrow[(size_t)(16 + i) * (HEADS_*KD_) + j * 16]; }
      v16h vb;
#pragma unroll
      for (int i = 0; i < 8; ++i) { vb[i] = vlo[i]; vb[8+i] = vhi[i]; }
      O[j] = __builtin_amdgcn_wmma_f32_16x16x32_f16(false, pa, false, vb, (short)0, O[j], false, false);
    }
  }
#pragma unroll
  for (int r = 0; r < 8; ++r) {
    float inv = 1.f / lrow[r];
    int qq = q0 + r + 8 * hsel;
    size_t base = ((size_t)(b * S_) + qq) * (HEADS_*KD_) + hd * KD_;
#pragma unroll
    for (int j = 0; j < 4; ++j) out16[base + j * 16 + l16] = (f16)(O[j][r] * inv);
  }
}

// ---------------------------------------------------------------------------
// Elementwise / small kernels
// ---------------------------------------------------------------------------
__global__ void k_cvt(const float* __restrict__ s, f16* __restrict__ d, int n) {
  int i = blockIdx.x * 256 + threadIdx.x;
  if (i < n) d[i] = (f16)s[i];
}

__global__ void k_posemb(float* __restrict__ pos) {   // [S,E]
  int idx = blockIdx.x * 256 + threadIdx.x;           // S*E
  int s = idx >> 8, e = idx & 255;
  float v = 0.f;
  if (e < 16)      { float f = 3.14159265358979f * exp2f(-(float)(1 + e));  v = sinf(s * f); }
  else if (e < 32) { float f = 3.14159265358979f * exp2f(-(float)(e - 15)); v = cosf(s * f); }
  pos[idx] = v;
}

__global__ void k_mask(const int* __restrict__ lengths, unsigned char* __restrict__ m) {
  int idx = blockIdx.x * 256 + threadIdx.x;
  if (idx < BS_) m[idx] = ((idx & (S_ - 1)) < lengths[idx >> 10]) ? 1 : 0;
}

__global__ void k_concat(const float* __restrict__ h, const int* __restrict__ pairs,
                         f16* __restrict__ cat) {     // cat16 [BS, 2E] = relu([pdata, h])
  size_t idx = (size_t)blockIdx.x * 256 + threadIdx.x;  // BS*2E
  int col = idx & 511;
  int row = (int)(idx >> 9);
  int b = row >> 10;
  float val;
  if (col < E_) {
    int pw = pairs[row];
    if (pw >= 0) {
      int ps = pw > S_ - 1 ? S_ - 1 : pw;
      val = h[((size_t)(b << 10) + ps) * E_ + col];
    } else val = 0.f;
  } else {
    val = h[(size_t)row * E_ + (col - E_)];
  }
  cat[idx] = (f16)(val > 0.f ? val : 0.f);
}

__global__ __launch_bounds__(256) void k_ln(const float* __restrict__ h,
    const float* __restrict__ gamma, const float* __restrict__ beta,
    const unsigned char* __restrict__ mask, float* __restrict__ o, f16* __restrict__ o16) {
  int wv = threadIdx.x >> 5, lane = threadIdx.x & 31;
  int row = blockIdx.x * 8 + wv;
  const float* hr = h + (size_t)row * H_;
  float s = 0.f, s2 = 0.f;
  for (int i = lane; i < H_; i += 32) { float v = hr[i]; s += v; s2 += v * v; }
  for (int off = 16; off > 0; off >>= 1) { s += __shfl_xor(s, off, 32); s2 += __shfl_xor(s2, off, 32); }
  float mu = s * (1.f / H_);
  float var = s2 * (1.f / H_) - mu * mu;
  float rs = rsqrtf(var + 1e-3f);
  float m = mask[row] ? 1.f : 0.f;
  for (int i = lane; i < H_; i += 32) {
    float v = (gamma[i] * (hr[i] - mu) * rs + beta[i]) * m;
    o[(size_t)row * H_ + i] = v;
    o16[(size_t)row * H_ + i] = (f16)v;
  }
}

__global__ void k_post_lstm(const float* __restrict__ hs, const unsigned char* __restrict__ mask,
                            float* __restrict__ h, f16* __restrict__ h16, int add) {
  size_t idx = (size_t)blockIdx.x * 256 + threadIdx.x;   // BS*H
  int row = (int)(idx >> 8);
  float v = hs[idx] * (mask[row] ? 1.f : 0.f);
  if (add) v += h[idx];
  h[idx] = v;
  h16[idx] = (f16)v;
}

// ---------------------------------------------------------------------------
extern "C" void kernel_launch(void* const* d_in, const int* in_sizes, int n_in,
                              void* d_out, int out_size, void* d_ws, size_t ws_size,
                              hipStream_t stream) {
  const float* x        = (const float*)d_in[0];
  const int*   lengths  = (const int*)  d_in[1];
  const int*   pairs    = (const int*)  d_in[2];
  const float* W_emb    = (const float*)d_in[3];
  const float* b_emb    = (const float*)d_in[4];
  const float* Wx       = (const float*)d_in[5];
  const float* Wh       = (const float*)d_in[6];
  const float* b_lstm   = (const float*)d_in[7];
  const float* W_pr     = (const float*)d_in[8];
  const float* b_pr     = (const float*)d_in[9];
  const float* ln_gamma = (const float*)d_in[10];
  const float* ln_beta  = (const float*)d_in[11];
  const float* Wqd      = (const float*)d_in[12];
  const float* bqd      = (const float*)d_in[13];
  const float* Wvd      = (const float*)d_in[14];
  const float* bvd      = (const float*)d_in[15];
  const float* Wq       = (const float*)d_in[16];
  const float* bq       = (const float*)d_in[17];
  const float* Wk       = (const float*)d_in[18];
  const float* bk       = (const float*)d_in[19];
  const float* Wv       = (const float*)d_in[20];
  const float* bv       = (const float*)d_in[21];
  const float* Wo       = (const float*)d_in[22];
  const float* bo       = (const float*)d_in[23];
  float* outp = (float*)d_out;

  // ---- workspace arena ----
  char* p = (char*)d_ws;
  auto alloc = [&](size_t bytes) -> char* { char* r = p; p += (bytes + 255) & ~(size_t)255; return r; };
  float* pos   = (float*)alloc(S_ * E_ * 4);
  unsigned char* maskb = (unsigned char*)alloc(BS_);
  f16*   x16   = (f16*)alloc((size_t)BS_ * FIN * 2);
  float* h     = (float*)alloc((size_t)BS_ * H_ * 4);
  f16*   h16   = (f16*)  alloc((size_t)BS_ * H_ * 2);
  f16*   cat16 = (f16*)  alloc((size_t)BS_ * 2 * E_ * 2);
  float* hln   = (float*)alloc((size_t)BS_ * H_ * 4);
  f16*   hln16 = (f16*)  alloc((size_t)BS_ * H_ * 2);
  float* zx    = (float*)alloc((size_t)BS_ * 4 * H_ * 4);
  float* hsbuf = (float*)alloc((size_t)BS_ * H_ * 4);
  f16*   q16   = (f16*)  alloc((size_t)BS_ * H_ * 2);
  f16*   v16   = (f16*)  alloc((size_t)BS_ * H_ * 2);
  f16*   qp16  = (f16*)  alloc((size_t)BS_ * H_ * 2);
  f16*   kt16  = (f16*)  alloc((size_t)B_ * H_ * S_ * 2);
  f16*   vp16  = (f16*)  alloc((size_t)BS_ * H_ * 2);
  f16*   ao16  = (f16*)  alloc((size_t)BS_ * H_ * 2);
  f16* w_emb16 = (f16*)alloc((size_t)FIN * E_ * 2);
  f16* wpr16   = (f16*)alloc((size_t)L_ * 2 * E_ * E_ * 2);
  f16* wx16    = (f16*)alloc((size_t)L_ * E_ * 4 * H_ * 2);
  f16* wh16    = (f16*)alloc((size_t)L_ * H_ * 4 * H_ * 2);
  f16* wqd16   = (f16*)alloc((size_t)L_ * H_ * H_ * 2);
  f16* wvd16   = (f16*)alloc((size_t)L_ * H_ * H_ * 2);
  f16* wq16    = (f16*)alloc((size_t)L_ * H_ * H_ * 2);
  f16* wk16    = (f16*)alloc((size_t)L_ * H_ * H_ * 2);
  f16* wv16    = (f16*)alloc((size_t)L_ * H_ * H_ * 2);
  f16* wo16    = (f16*)alloc((size_t)L_ * H_ * H_ * 2);

  auto cvt = [&](const float* s, f16* d, int n) {
    k_cvt<<<(n + 255) / 256, 256, 0, stream>>>(s, d, n);
  };
  auto gemm = [&](const f16* A, const f16* Bm, const float* bias, const float* posp,
                  const unsigned char* rm, const float* resid, float* C, f16* C16, f16* C16t,
                  int M, int N, int K, int act) {
    dim3 g(M / 64, N / 16);
    k_gemm<<<g, 128, 0, stream>>>(A, Bm, bias, posp, rm, resid, C, C16, C16t, M, N, K, act);
  };

  // ---- weight / input conversion + constants ----
  cvt(W_emb, w_emb16, FIN * E_);
  cvt(W_pr,  wpr16,   L_ * 2 * E_ * E_);
  cvt(Wx,    wx16,    L_ * E_ * 4 * H_);
  cvt(Wh,    wh16,    L_ * H_ * 4 * H_);
  cvt(Wqd,   wqd16,   L_ * H_ * H_);
  cvt(Wvd,   wvd16,   L_ * H_ * H_);
  cvt(Wq,    wq16,    L_ * H_ * H_);
  cvt(Wk,    wk16,    L_ * H_ * H_);
  cvt(Wv,    wv16,    L_ * H_ * H_);
  cvt(Wo,    wo16,    L_ * H_ * H_);
  cvt(x,     x16,     BS_ * FIN);
  k_posemb<<<(S_ * E_) / 256, 256, 0, stream>>>(pos);
  k_mask<<<(BS_ + 255) / 256, 256, 0, stream>>>(lengths, maskb);

  // ---- embedding: h = ((x @ W_emb + b) + pos) * m ----
  gemm(x16, w_emb16, b_emb, pos, maskb, nullptr, h, h16, nullptr, BS_, E_, FIN, 0);

  for (int i = 0; i < L_; ++i) {
    // pairing: cat = relu([gather(h), h]); h += (cat @ W_pr + b_pr) * m
    k_concat<<<(BS_ * 2 * E_) / 256, 256, 0, stream>>>(h, pairs, cat16);
    gemm(cat16, wpr16 + (size_t)i * 2 * E_ * E_, b_pr + i * E_, nullptr, maskb, h,
         h, h16, nullptr, BS_, E_, 2 * E_, 0);
    // layernorm (layer > 0)
    const f16* lstm_in16 = h16;
    if (i > 0) {
      k_ln<<<BS_ / 8, 256, 0, stream>>>(h, ln_gamma + (i - 1) * H_, ln_beta + (i - 1) * H_,
                                        maskb, hln, hln16);
      lstm_in16 = hln16;
    }
    // zx = lstm_in @ Wx + b_lstm
    gemm(lstm_in16, wx16 + (size_t)i * E_ * 4 * H_, b_lstm + i * 4 * H_, nullptr, nullptr,
         nullptr, zx, nullptr, nullptr, BS_, 4 * H_, E_, 0);
    // sequential recurrence (state in LDS, Wh from L2, zx tiles via TDM)
    k_lstm_scan<<<1, 512, 0, stream>>>(zx, wh16 + (size_t)i * H_ * 4 * H_, hsbuf);
    // h = hs*m (+ bypass for layer > 0)
    k_post_lstm<<<(BS_ * H_) / 256, 256, 0, stream>>>(hsbuf, maskb, h, h16, i > 0 ? 1 : 0);
    // q = relu(h@Wqd+bqd)*m ; v = relu(h@Wvd+bvd)*m   (f16 only)
    gemm(h16, wqd16 + (size_t)i * H_ * H_, bqd + i * H_, nullptr, maskb, nullptr,
         nullptr, q16, nullptr, BS_, H_, H_, 1);
    gemm(h16, wvd16 + (size_t)i * H_ * H_, bvd + i * H_, nullptr, maskb, nullptr,
         nullptr, v16, nullptr, BS_, H_, H_, 1);
    // head projections: Q normal f16, K transposed f16 [B, nk, S], V normal f16
    gemm(q16, wq16 + (size_t)i * H_ * H_, bq + i * H_, nullptr, nullptr, nullptr,
         nullptr, qp16, nullptr, BS_, H_, H_, 0);
    gemm(v16, wk16 + (size_t)i * H_ * H_, bk + i * H_, nullptr, nullptr, nullptr,
         nullptr, nullptr, kt16, BS_, H_, H_, 0);
    gemm(v16, wv16 + (size_t)i * H_ * H_, bv + i * H_, nullptr, nullptr, nullptr,
         nullptr, vp16, nullptr, BS_, H_, H_, 0);
    // WMMA flash attention
    k_attn_wmma<<<B_ * HEADS_ * (S_ / 64), 128, 0, stream>>>(qp16, kt16, vp16, lengths, ao16);
    // h = bypass + (attn @ Wo + bo) * m   (last layer writes d_out)
    float* dst = (i == L_ - 1) ? outp : h;
    gemm(ao16, wo16 + (size_t)i * H_ * H_, bo + i * H_, nullptr, maskb, h,
         dst, h16, nullptr, BS_, H_, H_, 0);
  }
}
